// Decoder_19172734009903
// MI455X (gfx1250) — matrix operand order repaired and checked by reading, not compiled, not run on hardware
//
#include <hip/hip_runtime.h>

// ---------------------------------------------------------------------------
// Problem constants (from reference)
// ---------------------------------------------------------------------------
#define B_    64
#define S_    1024
#define D_    256
#define H2_   512
#define F_    128
#define T_    64
#define L_    8
#define NSEG  128          // S / L
#define NROW  8192         // B * NSEG
#define DIN   1152         // H2 + D + 3F
#define G4    2048         // 4 * H2

typedef __bf16 bf16;
typedef __attribute__((ext_vector_type(16))) __bf16 v16bf;
typedef __attribute__((ext_vector_type(8)))  float  v8f;
typedef __attribute__((ext_vector_type(4)))  unsigned int v4u;
typedef __attribute__((ext_vector_type(8)))  unsigned int v8u;

// ---------------------------------------------------------------------------
// WMMA fragment helpers (wave32, layouts per CDNA5 ISA 7.12.2)
// ---------------------------------------------------------------------------
union AB32 { v16bf v; uint4 q[2]; };

// A fragment 16x32 bf16 from row-major src (ld in elements).
// lane l: m = l&15, kb = (l>>4)*16; elems 0..15 <- contiguous k0+kb ..
__device__ __forceinline__ v16bf load_a_frag(const bf16* src, int ld, int k0) {
  int lane = threadIdx.x & 31;
  int m  = lane & 15;
  int kb = (lane >> 4) << 4;
  const bf16* p = src + m * ld + k0 + kb;
  AB32 t;
  t.q[0] = *(const uint4*)(p);
  t.q[1] = *(const uint4*)(p + 8);
  return t.v;
}

// B fragment 32x16 bf16 from row-major W[N,K] (ldb = K).
// lane l: n = n0 + (l&15), kb = (l>>4)*16; elems 0..15 <- contiguous W[n][k0+kb ..]
__device__ __forceinline__ v16bf load_b_frag(const bf16* W, int ldb, int n0, int k0) {
  int lane = threadIdx.x & 31;
  int n  = n0 + (lane & 15);
  int kb = (lane >> 4) << 4;
  const bf16* p = W + n * ldb + k0 + kb;
  AB32 t;
  t.q[0] = *(const uint4*)(p);
  t.q[1] = *(const uint4*)(p + 8);
  return t.v;
}

__device__ __forceinline__ v8f wmma_bf16(v16bf a, v16bf b, v8f c) {
  // (neg_a, A, neg_b, B, c_mod, C, reuse_a, reuse_b)
  return __builtin_amdgcn_wmma_f32_16x16x32_bf16(false, a, false, b, (short)0, c,
                                                 false, false);
}

// D fragment 16x16 f32 store into row-major dst (ld elements).
// lane l: n = l&15, rows m = (l>>4)*8 + r
__device__ __forceinline__ void store_d(float* dst, int ld, v8f d) {
  int lane = threadIdx.x & 31;
  int n  = lane & 15;
  int mb = (lane >> 4) << 3;
#pragma unroll
  for (int r = 0; r < 8; ++r) dst[(mb + r) * ld + n] = d[r];
}

// ---------------------------------------------------------------------------
// Tensor Data Mover: 2D tile load Global -> LDS (CDNA5 ISA ch.8, D# groups 0/1)
// data_size = 2 bytes (bf16). Issued by one wave; completion via TENSORcnt.
// ---------------------------------------------------------------------------
__device__ __forceinline__ void tdm_load_2d(const void* gaddr, unsigned lds_off,
                                            unsigned tile_d0, unsigned tile_d1,
                                            unsigned stride_e,
                                            unsigned tensor_d0, unsigned tensor_d1) {
  unsigned long long ga = (unsigned long long)(uintptr_t)gaddr;
  // Group 0: [1:0]=count=1, [63:32]=lds_addr, [120:64]=global_addr, [127:126]=type=2
  v4u g0 = { 1u,
             lds_off,
             (unsigned)ga,
             (((unsigned)(ga >> 32)) & 0x01FFFFFFu) | (2u << 30) };
  // Group 1: [17:16]=data_size(2B), [79:48]=tensor_dim0, [111:80]=tensor_dim1,
  //          [127:112]=tile_dim0, [143:128]=tile_dim1, [207:160]=dim0_stride
  v8u g1 = { 0x00010000u,
             (tensor_d0 & 0xFFFFu) << 16,
             ((tensor_d0 >> 16) & 0xFFFFu) | ((tensor_d1 & 0xFFFFu) << 16),
             ((tensor_d1 >> 16) & 0xFFFFu) | (tile_d0 << 16),
             tile_d1,
             stride_e,
             0u,
             0u };
  asm volatile("tensor_load_to_lds %0, %1" : : "s"(g0), "s"(g1) : "memory");
}

__device__ __forceinline__ unsigned lds_addr_of(const void* p) {
  return (unsigned)(uintptr_t)p;
}

// ---------------------------------------------------------------------------
// Utility kernels
// ---------------------------------------------------------------------------
__global__ void cvt_bf16_kernel(const float* __restrict__ src, bf16* __restrict__ dst, int n) {
  int i = blockIdx.x * blockDim.x + threadIdx.x;
  if (i < n) dst[i] = (bf16)src[i];
}

// Embedding gather -> wf (bf16) + per-segment mean -> dec_in[:, 0:256]
__global__ void gather_embed_kernel(const int* __restrict__ word_ids,
                                    const float* __restrict__ embed,
                                    bf16* __restrict__ wf,
                                    bf16* __restrict__ dec_in) {
  int n = blockIdx.x;            // segment row, n = b*NSEG + seg
  int c = threadIdx.x;           // column 0..255
  float sum = 0.f;
#pragma unroll
  for (int r = 0; r < L_; ++r) {
    int srow = n * L_ + r;       // == b*S + seg*L + r
    int id = word_ids[srow];
    float e = embed[id * D_ + c];
    wf[srow * D_ + c] = (bf16)e;
    sum += e;
  }
  dec_in[n * DIN + c] = (bf16)(sum * 0.125f);
}

// Per-segment mean of encoder outputs -> dec_in[:, 256:768]
__global__ void seg_mean_enc_kernel(const float* __restrict__ enc,
                                    bf16* __restrict__ dec_in) {
  int n = blockIdx.x;
#pragma unroll
  for (int jj = 0; jj < 2; ++jj) {
    int j = threadIdx.x + jj * 256;   // 0..511
    float sum = 0.f;
#pragma unroll
    for (int r = 0; r < L_; ++r) sum += enc[(n * L_ + r) * H2_ + j];
    dec_in[n * DIN + D_ + j] = (bf16)(sum * 0.125f);
  }
}

// h0 = concat(enc[:, -1, :256], enc[:, 0, 256:]); c0 = 0
__global__ void init_state_kernel(const float* __restrict__ enc,
                                  bf16* __restrict__ hbf,
                                  float* __restrict__ c) {
  int idx = blockIdx.x * blockDim.x + threadIdx.x;   // 0 .. 64*512-1
  if (idx >= B_ * H2_) return;
  int b = idx >> 9, j = idx & 511;
  float v = (j < 256) ? enc[(b * S_ + (S_ - 1)) * H2_ + j]
                      : enc[(b * S_ + 0) * H2_ + j];
  hbf[idx] = (bf16)v;
  c[idx] = 0.f;
}

// ---------------------------------------------------------------------------
// Char-CNN via WMMA: one block = 2 segments (16 A rows), 8 waves = 8 filter tiles
// ---------------------------------------------------------------------------
template <int FS>
__device__ __forceinline__ void conv_do(const bf16* xt, const bf16* W,
                                        const float* __restrict__ bias,
                                        bf16* __restrict__ dec_in,
                                        int pair, int f0, int colbase) {
  v8f acc = {};
#pragma unroll
  for (int ki = 0; ki < FS * 8; ++ki) {
    int k0  = ki * 32;
    int tap = k0 >> 8;          // which shifted row (filter tap)
    int d0  = k0 & 255;         // channel offset within tap
    v16bf a = load_a_frag(xt + tap * D_ + d0, D_, 0);
    v16bf b = load_b_frag(W, FS * D_, f0, k0);
    acc = wmma_bf16(a, b, acc);
  }
  // D layout: lanes 0-15 hold seg0 rows t=0..7 in acc[0..7]; lanes 16-31 seg1.
  int lane = threadIdx.x & 31;
  int f    = f0 + (lane & 15);
  int seg  = lane >> 4;
  float bv = bias[f];
  float mx = 0.f;               // ReLU => max >= 0
#pragma unroll
  for (int r = 0; r < 9 - FS; ++r)        // valid positions t = 0 .. L-FS
    mx = fmaxf(mx, fmaxf(acc[r] + bv, 0.f));
  int n = pair * 2 + seg;
  dec_in[n * DIN + colbase + f] = (bf16)mx;
}

__global__ void __launch_bounds__(256)
conv_kernel(const bf16* __restrict__ wf,
            const bf16* __restrict__ w1, const float* __restrict__ b1,
            const bf16* __restrict__ w2, const float* __restrict__ b2,
            const bf16* __restrict__ w3, const float* __restrict__ b3,
            bf16* __restrict__ dec_in) {
  __shared__ __align__(16) bf16 xt[18 * D_];   // 16 rows (2 segs) + 2 halo rows
  int pair = blockIdx.x;                       // 0 .. 4095
  // cooperative load: 18 rows x 256 bf16 = 576 16B-chunks
  for (int i = threadIdx.x; i < 18 * 32; i += 256) {
    int row = i >> 5;
    int c8  = (i & 31) << 3;
    int g   = pair * 16 + row;
    if (g > NROW * L_ - 1) g = NROW * L_ - 1;  // halo clamp (pooled-out anyway)
    *(uint4*)(xt + row * D_ + c8) = *(const uint4*)(wf + g * D_ + c8);
  }
  __syncthreads();
  int f0 = (threadIdx.x >> 5) * 16;            // wave -> filter tile
  conv_do<1>(xt, w1, b1, dec_in, pair, f0, D_ + H2_);
  conv_do<2>(xt, w2, b2, dec_in, pair, f0, D_ + H2_ + F_);
  conv_do<3>(xt, w3, b3, dec_in, pair, f0, D_ + H2_ + 2 * F_);
}

// ---------------------------------------------------------------------------
// Big GEMM: X_ih[8192,2048] = dec_in[8192,1152](bf16) x W_ih[2048,1152]^T
// WG tile 64x128, 8 waves. A staged via TDM into double-buffered LDS,
// K-chunks of 128, overlap tracked with TENSORcnt.
// ---------------------------------------------------------------------------
__global__ void __launch_bounds__(256)
gemm_xih_kernel(const bf16* __restrict__ decin, const bf16* __restrict__ Wih,
                float* __restrict__ Xih) {
  __shared__ __align__(16) bf16 at[2][64 * 128];
  int nb = blockIdx.x * 128;
  int mb = blockIdx.y * 64;
  int w = threadIdx.x >> 5;
  int m_tile = w & 3;
  int n_half = w >> 2;
  bool w0 = (threadIdx.x < 32);
  const bf16* base = decin + mb * DIN;
  v8f acc[4] = {{}, {}, {}, {}};
  if (w0)
    tdm_load_2d(base, lds_addr_of(at[0]), 128, 64, DIN, DIN, NROW - mb);
  for (int kc = 0; kc < 9; ++kc) {           // 1152 / 128
    if (w0) {
      if (kc + 1 < 9) {
        tdm_load_2d(base + (kc + 1) * 128, lds_addr_of(at[(kc + 1) & 1]),
                    128, 64, DIN, DIN, NROW - mb);
        __builtin_amdgcn_s_wait_tensorcnt(1);   // current buffer landed
      } else {
        __builtin_amdgcn_s_wait_tensorcnt(0);
      }
    }
    __syncthreads();
    const bf16* atc = at[kc & 1];
    if (kc < 8)
      __builtin_prefetch(Wih + (nb + n_half * 64 + (threadIdx.x & 63)) * DIN +
                             (kc + 1) * 128, 0, 1);
#pragma unroll
    for (int ki = 0; ki < 4; ++ki) {
      v16bf a = load_a_frag(atc + m_tile * 16 * 128, 128, ki * 32);
#pragma unroll
      for (int j = 0; j < 4; ++j) {
        v16bf b = load_b_frag(Wih, DIN, nb + n_half * 64 + j * 16, kc * 128 + ki * 32);
        acc[j] = wmma_bf16(a, b, acc[j]);
      }
    }
    __syncthreads();
  }
#pragma unroll
  for (int j = 0; j < 4; ++j)
    store_d(Xih + (mb + m_tile * 16) * G4 + nb + n_half * 64 + j * 16, G4, acc[j]);
}

// ---------------------------------------------------------------------------
// Per-step recurrent GEMM: gates[64,2048] = X_ih[b,step,:] + h(bf16) x W_hh^T
// 16 WGs (N blocks of 128). A (h) staged via TDM, double-buffered, K=512.
// ---------------------------------------------------------------------------
__global__ void __launch_bounds__(256)
gemm_gates_kernel(const bf16* __restrict__ hbf, const bf16* __restrict__ Whh,
                  const float* __restrict__ Xih, float* __restrict__ gates,
                  int step) {
  __shared__ __align__(16) bf16 at[2][64 * 128];
  int nb = blockIdx.x * 128;
  int w = threadIdx.x >> 5;
  int m_tile = w & 3;
  int n_half = w >> 2;
  bool w0 = (threadIdx.x < 32);
  v8f acc[4] = {{}, {}, {}, {}};
  if (w0)
    tdm_load_2d(hbf, lds_addr_of(at[0]), 128, 64, H2_, H2_, B_);
  for (int kc = 0; kc < 4; ++kc) {           // 512 / 128
    if (w0) {
      if (kc + 1 < 4) {
        tdm_load_2d(hbf + (kc + 1) * 128, lds_addr_of(at[(kc + 1) & 1]),
                    128, 64, H2_, H2_, B_);
        __builtin_amdgcn_s_wait_tensorcnt(1);
      } else {
        __builtin_amdgcn_s_wait_tensorcnt(0);
      }
    }
    __syncthreads();
    const bf16* atc = at[kc & 1];
#pragma unroll
    for (int ki = 0; ki < 4; ++ki) {
      v16bf a = load_a_frag(atc + m_tile * 16 * 128, 128, ki * 32);
#pragma unroll
      for (int j = 0; j < 4; ++j) {
        v16bf b = load_b_frag(Whh, H2_, nb + n_half * 64 + j * 16, kc * 128 + ki * 32);
        acc[j] = wmma_bf16(a, b, acc[j]);
      }
    }
    __syncthreads();
  }
  int lane = threadIdx.x & 31;
  int nloc = lane & 15;
  int mbk  = (lane >> 4) << 3;
#pragma unroll
  for (int j = 0; j < 4; ++j) {
    int col = nb + n_half * 64 + j * 16 + nloc;
#pragma unroll
    for (int r = 0; r < 8; ++r) {
      int bb = m_tile * 16 + mbk + r;        // batch row
      gates[bb * G4 + col] = acc[j][r] + Xih[(bb * NSEG + step) * G4 + col];
    }
  }
}

// ---------------------------------------------------------------------------
// LSTM elementwise update
// ---------------------------------------------------------------------------
__device__ __forceinline__ float sigm(float x) { return 1.f / (1.f + __expf(-x)); }

__global__ void lstm_cell_kernel(const float* __restrict__ gates,
                                 float* __restrict__ c, bf16* __restrict__ hbf) {
  int idx = blockIdx.x * blockDim.x + threadIdx.x;   // 0 .. 64*512-1
  if (idx >= B_ * H2_) return;
  int b = idx >> 9, j = idx & 511;
  float gi = gates[b * G4 + j];
  float gf = gates[b * G4 + H2_ + j];
  float gg = gates[b * G4 + 2 * H2_ + j];
  float go = gates[b * G4 + 3 * H2_ + j];
  float cn = sigm(gf) * c[idx] + sigm(gi) * tanhf(gg);
  c[idx] = cn;
  hbf[idx] = (bf16)(sigm(go) * tanhf(cn));
}

// ---------------------------------------------------------------------------
// Logits + log_softmax: logits[64,64] = h x W_fc^T, then row log-softmax.
// One WG of 512 threads = 16 waves = 4x4 tile grid over the 64x64 output.
// ---------------------------------------------------------------------------
__global__ void __launch_bounds__(512)
logits_kernel(const bf16* __restrict__ hbf, const bf16* __restrict__ Wfc,
              float* __restrict__ out, int step) {
  __shared__ __align__(16) float lg[T_ * T_];
  int w = threadIdx.x >> 5;
  int m_tile = w >> 2;
  int n_tile = w & 3;
  v8f acc = {};
#pragma unroll
  for (int ki = 0; ki < 16; ++ki) {          // K = 512
    v16bf a = load_a_frag(hbf + m_tile * 16 * H2_, H2_, ki * 32);
    v16bf b = load_b_frag(Wfc, H2_, n_tile * 16, ki * 32);
    acc = wmma_bf16(a, b, acc);
  }
  store_d(lg + m_tile * 16 * T_ + n_tile * 16, T_, acc);
  __syncthreads();
  int t = threadIdx.x;
  if (t < T_) {                              // one thread per batch row
    float mx = -1e30f;
#pragma unroll 8
    for (int cix = 0; cix < T_; ++cix) mx = fmaxf(mx, lg[t * T_ + cix]);
    float s = 0.f;
#pragma unroll 8
    for (int cix = 0; cix < T_; ++cix) s += __expf(lg[t * T_ + cix] - mx);
    float lse = mx + __logf(s);
    float* orow = out + (t * NSEG + step) * T_;   // row = b*NSEG + step
#pragma unroll 8
    for (int cix = 0; cix < T_; ++cix) orow[cix] = lg[t * T_ + cix] - lse;
  }
}

// ---------------------------------------------------------------------------
// Host-side orchestration
// ---------------------------------------------------------------------------
extern "C" void kernel_launch(void* const* d_in, const int* in_sizes, int n_in,
                              void* d_out, int out_size, void* d_ws, size_t ws_size,
                              hipStream_t stream) {
  const int*   word_ids = (const int*)  d_in[0];
  const float* embed    = (const float*)d_in[1];
  const float* enc      = (const float*)d_in[2];
  const float* wconv1   = (const float*)d_in[3];
  const float* bconv1   = (const float*)d_in[4];
  const float* wconv2   = (const float*)d_in[5];
  const float* bconv2   = (const float*)d_in[6];
  const float* wconv3   = (const float*)d_in[7];
  const float* bconv3   = (const float*)d_in[8];
  const float* W_ih     = (const float*)d_in[9];
  const float* W_hh     = (const float*)d_in[10];
  const float* W_fc     = (const float*)d_in[11];
  float* out = (float*)d_out;

  // workspace carve-up (256B aligned regions)
  char* p = (char*)d_ws;
  auto alloc = [&](size_t bytes) {
    char* r = p;
    p += (bytes + 255) & ~(size_t)255;
    return r;
  };
  bf16*  ws_wf    = (bf16*) alloc((size_t)B_ * S_ * D_ * 2);     // 32 MB
  bf16*  ws_decin = (bf16*) alloc((size_t)NROW * DIN * 2);       // 18 MB
  bf16*  ws_Wih   = (bf16*) alloc((size_t)G4 * DIN * 2);         // 4.6 MB
  bf16*  ws_Whh   = (bf16*) alloc((size_t)G4 * H2_ * 2);         // 2 MB
  bf16*  ws_Wfc   = (bf16*) alloc((size_t)T_ * H2_ * 2);
  bf16*  ws_w1    = (bf16*) alloc((size_t)F_ * D_ * 2);
  bf16*  ws_w2    = (bf16*) alloc((size_t)F_ * 2 * D_ * 2);
  bf16*  ws_w3    = (bf16*) alloc((size_t)F_ * 3 * D_ * 2);
  float* ws_Xih   = (float*)alloc((size_t)NROW * G4 * 4);        // 64 MB
  float* ws_gates = (float*)alloc((size_t)B_ * G4 * 4);
  bf16*  ws_h     = (bf16*) alloc((size_t)B_ * H2_ * 2);
  float* ws_c     = (float*)alloc((size_t)B_ * H2_ * 4);

  auto cvt = [&](const float* src, bf16* dst, int n) {
    cvt_bf16_kernel<<<(n + 255) / 256, 256, 0, stream>>>(src, dst, n);
  };
  // weight conversions to bf16
  cvt(W_ih,   ws_Wih, G4 * DIN);
  cvt(W_hh,   ws_Whh, G4 * H2_);
  cvt(W_fc,   ws_Wfc, T_ * H2_);
  cvt(wconv1, ws_w1,  F_ * D_);
  cvt(wconv2, ws_w2,  F_ * 2 * D_);
  cvt(wconv3, ws_w3,  F_ * 3 * D_);

  // features
  gather_embed_kernel<<<NROW, 256, 0, stream>>>(word_ids, embed, ws_wf, ws_decin);
  seg_mean_enc_kernel<<<NROW, 256, 0, stream>>>(enc, ws_decin);
  conv_kernel<<<NROW / 2, 256, 0, stream>>>(ws_wf, ws_w1, bconv1, ws_w2, bconv2,
                                            ws_w3, bconv3, ws_decin);

  // hoisted input-path GEMM: X_ih = dec_in @ W_ih^T
  {
    dim3 grid(G4 / 128, NROW / 64);
    gemm_xih_kernel<<<grid, 256, 0, stream>>>(ws_decin, ws_Wih, ws_Xih);
  }

  // decoder init
  init_state_kernel<<<(B_ * H2_ + 255) / 256, 256, 0, stream>>>(enc, ws_h, ws_c);

  // sequential LSTM scan (graph-captured; launch latency amortized on replay)
  for (int step = 0; step < NSEG; ++step) {
    gemm_gates_kernel<<<G4 / 128, 256, 0, stream>>>(ws_h, ws_Whh, ws_Xih,
                                                    ws_gates, step);
    lstm_cell_kernel<<<(B_ * H2_ + 255) / 256, 256, 0, stream>>>(ws_gates, ws_c, ws_h);
    logits_kernel<<<1, 512, 0, stream>>>(ws_h, ws_Wfc, out, step);
  }
}
